// Recursive_Gauss_LSTM_7868380086569
// MI455X (gfx1250) — compile-verified, matrix operand order checked by reading, not run on hardware
//
#include <hip/hip_runtime.h>

// ---------------------------------------------------------------------------
// Gaussian LSTM for MI455X (gfx1250, wave32, WMMA bf16)
//   B=64, T=512, D=HS=256, 4HS=1024, 2HS=512
// Phase 1: transpose/convert weights fp32 -> bf16 [N][K]
// Phase 2: xW = X @ W_ih  (massively parallel WMMA GEMM, fp32 out to ws)
// Phase 3: persistent 1-workgroup recurrence; h/c/pre_h/pre_c/p in LDS;
//          each weight byte read exactly once per step from L2.
// ---------------------------------------------------------------------------

typedef __attribute__((ext_vector_type(16))) __bf16 v16bf;
typedef __attribute__((ext_vector_type(8)))  __bf16 v8bf;
typedef __attribute__((ext_vector_type(8)))  float  v8f;
typedef __attribute__((ext_vector_type(4)))  float  v4f;

#define HSZ   256
#define BSZ   64
#define TLEN  512
#define NGATE 1024   // 4*HS
#define NREP  512    // 2*HS
#define MROWS (BSZ * TLEN)   // 32768 rows of X / xW

static __device__ __forceinline__ v8f wmma_bf16(v16bf a, v16bf b, v8f c) {
    return __builtin_amdgcn_wmma_f32_16x16x32_bf16(
        /*neg_a=*/false, a, /*neg_b=*/false, b,
        /*c_mod=*/(short)0, c, /*reuse_a=*/false, /*reuse_b=*/false);
}

// Load a 16-element bf16 fragment for this lane.
// p points at (row_for_this_lane, kb + (lane>=16 ? 8 : 0)); the second
// contiguous 8-element group sits 16 elements further along K.
static __device__ __forceinline__ v16bf load_frag_bf16(const __bf16* p) {
    v8bf lo = *(const v8bf*)(p);
    v8bf hi = *(const v8bf*)(p + 16);
    return __builtin_shufflevector(lo, hi,
        0, 1, 2, 3, 4, 5, 6, 7, 8, 9, 10, 11, 12, 13, 14, 15);
}

// Same fragment but converting from fp32 on the fly (for X).
static __device__ __forceinline__ v16bf load_frag_f32(const float* p) {
    v4f f0 = *(const v4f*)(p);
    v4f f1 = *(const v4f*)(p + 4);
    v4f f2 = *(const v4f*)(p + 16);
    v4f f3 = *(const v4f*)(p + 20);
    v16bf a;
#pragma unroll
    for (int i = 0; i < 4; ++i) {
        a[i]      = (__bf16)f0[i];
        a[4 + i]  = (__bf16)f1[i];
        a[8 + i]  = (__bf16)f2[i];
        a[12 + i] = (__bf16)f3[i];
    }
    return a;
}

static __device__ __forceinline__ float fast_sigmoid(float x) {
    return 1.0f / (1.0f + __expf(-x));
}
static __device__ __forceinline__ float fast_tanh(float x) {
    return 1.0f - 2.0f / (__expf(2.0f * x) + 1.0f);
}
static __device__ __forceinline__ float softplus_f(float x) {
    if (x > 20.0f)  return x;
    if (x < -20.0f) return __expf(x);
    return __logf(1.0f + __expf(x));
}

// ---------------------------------------------------------------------------
// Prep: out[n][k] (bf16) = in[k][n] (fp32);  in is [K][N]
// ---------------------------------------------------------------------------
__global__ void transpose_cvt_kernel(const float* __restrict__ in,
                                     __bf16* __restrict__ out,
                                     int K, int N) {
    int idx = blockIdx.x * blockDim.x + threadIdx.x;
    if (idx >= K * N) return;
    int n = idx / K;
    int k = idx - n * K;
    out[(size_t)n * K + k] = (__bf16)in[(size_t)k * N + n];
}

// ---------------------------------------------------------------------------
// Phase 2: xW[m][n] = sum_k X[m][k] * W_ih[k][n]
// One wave computes a 16x64 output strip (4 WMMA tiles sharing A).
// ---------------------------------------------------------------------------
__global__ void __launch_bounds__(256)
gemm_xw_kernel(const float* __restrict__ X,    // [32768][256] fp32
               const __bf16* __restrict__ Wt,  // [1024][256]  bf16 (transposed)
               float* __restrict__ xW) {       // [32768][1024] fp32
    const int wave = blockIdx.x * (blockDim.x >> 5) + (threadIdx.x >> 5);
    const int mb = wave >> 4;   // row-block 0..2047
    const int ng = wave & 15;   // 64-wide column group 0..15
    const int ln  = threadIdx.x & 31;
    const int sel = ln >> 4;    // lane half -> K sub-select
    const int l15 = ln & 15;

    const float* aBase = X + (size_t)(mb * 16 + l15) * HSZ + sel * 8;

    v8f acc[4] = {v8f{}, v8f{}, v8f{}, v8f{}};
#pragma unroll
    for (int kb = 0; kb < HSZ; kb += 32) {
        v16bf a = load_frag_f32(aBase + kb);
#pragma unroll
        for (int nt = 0; nt < 4; ++nt) {
            int ncol = ng * 64 + nt * 16 + l15;
            v16bf b = load_frag_bf16(Wt + (size_t)ncol * HSZ + kb + sel * 8);
            acc[nt] = wmma_bf16(a, b, acc[nt]);
        }
    }
    const int mrow0 = mb * 16 + sel * 8;   // C/D layout: lane>=16 -> M += 8
#pragma unroll
    for (int nt = 0; nt < 4; ++nt) {
        int ncol = ng * 64 + nt * 16 + l15;
#pragma unroll
        for (int r = 0; r < 8; ++r)
            xW[(size_t)(mrow0 + r) * NGATE + ncol] = acc[nt][r];
    }
}

// ---------------------------------------------------------------------------
// Phase 3: persistent single-workgroup recurrence (1024 threads = 32 waves).
// LDS: h(bf16 32K) | pre_h(bf16 32K) | pre_c(bf16 32K) | c(f32 64K) | p(f32 64K)
//      = 224 KB dynamic (WGP has 320 KB).
//
// Phase G decomposition: wave w -> column-block cb = w&15 (16 cols of HS),
//   gate-pair = w>>4 (pair0: {i,g}, pair1: {f,o}); each wave covers all 4
//   row-blocks so every W_hh column fragment is loaded exactly once/step.
//   pair0 stages p = sigmoid(i)*tanh(g) in LDS; after a barrier pair1 (whose
//   raw f/o accumulators persist in registers) finishes the cell update.
// Phase R decomposition: wave w -> path = w>>4 (h or c), col-pair cp = w&15;
//   computes the mu tile and std tile across all 4 row-blocks, so reparam
//   weights are read exactly once/step and mu/std stay lane-aligned.
// ---------------------------------------------------------------------------
__global__ void __launch_bounds__(1024, 1)
lstm_recurrent_kernel(const float* __restrict__ xW,     // [B*T][1024] (row = b*T+t)
                      const float* __restrict__ bias,   // [1024]
                      const __bf16* __restrict__ WhhT,  // [1024][256]
                      const __bf16* __restrict__ HrepT, // [512][256]
                      const __bf16* __restrict__ CrepT, // [512][256]
                      const float* __restrict__ eps_h,  // [T][B][HS]
                      const float* __restrict__ eps_c,  // [T][B][HS]
                      float* __restrict__ out) {
    extern __shared__ char smem[];
    __bf16* h_bf  = (__bf16*)smem;                 // 64*256 bf16
    __bf16* ph_bf = h_bf + BSZ * HSZ;              // pre_h
    __bf16* pc_bf = ph_bf + BSZ * HSZ;             // pre_c
    float*  c_f   = (float*)(pc_bf + BSZ * HSZ);   // 64*256 f32
    float*  p_f   = c_f + BSZ * HSZ;               // sigmoid(i)*tanh(g), f32

    const int tid = threadIdx.x;
    const int wv  = tid >> 5;         // wave 0..31
    const int ln  = tid & 31;
    const int sel = ln >> 4;
    const int l15 = ln & 15;

    for (int i = tid; i < BSZ * HSZ; i += blockDim.x) {
        h_bf[i] = (__bf16)0.0f;
        c_f[i]  = 0.0f;
    }
    __syncthreads();

    const size_t S = (size_t)BSZ * TLEN * HSZ;   // one output plane

    // Phase-G wave mapping (constant over t)
    const int cb   = wv & 15;        // 16-column slab within HS
    const int pair = wv >> 4;        // 0: gates {i,g}; 1: gates {f,o}
    const int gA   = (pair == 0) ? 0 : 1;   // i or f
    const int gB   = (pair == 0) ? 2 : 3;   // g or o
    // Phase-R wave mapping
    const int isC = wv >> 4;         // 0: h-path, 1: c-path
    const int cp  = wv & 15;

    for (int t = 0; t < TLEN; ++t) {
        // ---------------- Phase G: gates = xW[t] + bias + h @ W_hh ----------
        v8f acc[2][4];   // [gate-of-pair][row-block]
        {
            const int ncA = gA * HSZ + cb * 16 + l15;
            const int ncB = gB * HSZ + cb * 16 + l15;
            const float bA = bias[ncA];
            const float bB = bias[ncB];
#pragma unroll
            for (int rb = 0; rb < 4; ++rb) {
                const int mrow0 = rb * 16 + sel * 8;
#pragma unroll
                for (int r = 0; r < 8; ++r) {
                    const size_t rowoff = ((size_t)(mrow0 + r) * TLEN + t) * NGATE;
                    acc[0][rb][r] = xW[rowoff + ncA] + bA;
                    acc[1][rb][r] = xW[rowoff + ncB] + bB;
                }
                // prefetch next step's xW rows for this wave (global_prefetch)
                if (t + 1 < TLEN) {
                    const size_t nrow = ((size_t)(mrow0) * TLEN + t + 1) * NGATE;
                    __builtin_prefetch(&xW[nrow + ncA], 0, 1);
                    __builtin_prefetch(&xW[nrow + ncB], 0, 1);
                }
            }
#pragma unroll
            for (int kb = 0; kb < HSZ; kb += 32) {
                v16bf a[4];
#pragma unroll
                for (int rb = 0; rb < 4; ++rb)
                    a[rb] = load_frag_bf16(h_bf + (size_t)(rb * 16 + l15) * HSZ + kb + sel * 8);
                v16bf bfA = load_frag_bf16(WhhT + (size_t)ncA * HSZ + kb + sel * 8);
                v16bf bfB = load_frag_bf16(WhhT + (size_t)ncB * HSZ + kb + sel * 8);
#pragma unroll
                for (int rb = 0; rb < 4; ++rb) {
                    acc[0][rb] = wmma_bf16(a[rb], bfA, acc[0][rb]);
                    acc[1][rb] = wmma_bf16(a[rb], bfB, acc[1][rb]);
                }
            }
        }
        const int j = cb * 16 + l15;
        if (pair == 0) {
            // p = sigmoid(i) * tanh(g)  -> LDS
#pragma unroll
            for (int rb = 0; rb < 4; ++rb) {
                const int mrow0 = rb * 16 + sel * 8;
#pragma unroll
                for (int r = 0; r < 8; ++r) {
                    const int b = mrow0 + r;
                    p_f[b * HSZ + j] = fast_sigmoid(acc[0][rb][r]) * fast_tanh(acc[1][rb][r]);
                }
            }
        }
        __syncthreads();   // p visible

        if (pair == 1) {
            // cell update: pre_c = sigmoid(f)*c + p ; pre_h = sigmoid(o)*tanh(c)
#pragma unroll
            for (int rb = 0; rb < 4; ++rb) {
                const int mrow0 = rb * 16 + sel * 8;
#pragma unroll
                for (int r = 0; r < 8; ++r) {
                    const int b = mrow0 + r;
                    const float ft = fast_sigmoid(acc[0][rb][r]);
                    const float ot = fast_sigmoid(acc[1][rb][r]);
                    const float cold  = c_f[b * HSZ + j];
                    const float pre_c = ft * cold + p_f[b * HSZ + j];
                    const float pre_h = ot * fast_tanh(cold);
                    ph_bf[b * HSZ + j] = (__bf16)pre_h;
                    pc_bf[b * HSZ + j] = (__bf16)pre_c;
                }
            }
        }
        __syncthreads();   // pre_h / pre_c visible

        // ---------------- Phase R: reparam GEMMs + Gaussian sample ----------
        {
            const __bf16* A  = isC ? pc_bf : ph_bf;
            const __bf16* Rt = isC ? CrepT : HrepT;
            const int nmu = cp * 16 + l15;         // mu column
            const int nsp = HSZ + cp * 16 + l15;   // std-param column

            v8f amu[4] = {v8f{}, v8f{}, v8f{}, v8f{}};
            v8f asp[4] = {v8f{}, v8f{}, v8f{}, v8f{}};
#pragma unroll
            for (int kb = 0; kb < HSZ; kb += 32) {
                v16bf a[4];
#pragma unroll
                for (int rb = 0; rb < 4; ++rb)
                    a[rb] = load_frag_bf16(A + (size_t)(rb * 16 + l15) * HSZ + kb + sel * 8);
                v16bf b0 = load_frag_bf16(Rt + (size_t)nmu * HSZ + kb + sel * 8);
                v16bf b1 = load_frag_bf16(Rt + (size_t)nsp * HSZ + kb + sel * 8);
#pragma unroll
                for (int rb = 0; rb < 4; ++rb) {
                    amu[rb] = wmma_bf16(a[rb], b0, amu[rb]);
                    asp[rb] = wmma_bf16(a[rb], b1, asp[rb]);
                }
            }
            const int jj = cp * 16 + l15;
            const float* eps = isC ? eps_c : eps_h;
#pragma unroll
            for (int rb = 0; rb < 4; ++rb) {
                const int mrow0 = rb * 16 + sel * 8;
#pragma unroll
                for (int r = 0; r < 8; ++r) {
                    const int b = mrow0 + r;
                    const float mu = fminf(fmaxf(amu[rb][r], 1e-6f), 1e6f);
                    const float sd = fmaxf(softplus_f(asp[rb][r]), 1e-6f);
                    const float e  = eps[((size_t)t * BSZ + b) * HSZ + jj];
                    const float val = mu + e * sd;
                    const size_t oidx = (size_t)b * (TLEN * HSZ) + (size_t)t * HSZ + jj;
                    if (!isC) {
                        out[oidx]         = val;   // hidden_seq
                        out[2 * S + oidx] = mu;    // h_mus
                        out[3 * S + oidx] = sd;    // h_stds
                        h_bf[b * HSZ + jj] = (__bf16)val;
                        if (t == TLEN - 1)
                            out[6 * S + (size_t)b * HSZ + jj] = val;   // h_t
                    } else {
                        out[S + oidx]     = val;   // c_seq
                        out[4 * S + oidx] = mu;    // c_mus
                        out[5 * S + oidx] = sd;    // c_stds
                        c_f[b * HSZ + jj] = val;
                        if (t == TLEN - 1)
                            out[6 * S + (size_t)BSZ * HSZ + (size_t)b * HSZ + jj] = val;  // c_t
                    }
                }
            }
        }
        __syncthreads();   // h/c updated before next step's Phase G
    }
}

// ---------------------------------------------------------------------------
// Host launcher
// Inputs: x, weight_ih, weight_hh, bias, h_reparam, c_reparam, eps_h, eps_c
// ---------------------------------------------------------------------------
extern "C" void kernel_launch(void* const* d_in, const int* in_sizes, int n_in,
                              void* d_out, int out_size, void* d_ws, size_t ws_size,
                              hipStream_t stream) {
    const float* x     = (const float*)d_in[0];   // [64][512][256]
    const float* w_ih  = (const float*)d_in[1];   // [256][1024]
    const float* w_hh  = (const float*)d_in[2];   // [256][1024]
    const float* bias  = (const float*)d_in[3];   // [1024]
    const float* h_rep = (const float*)d_in[4];   // [256][512]
    const float* c_rep = (const float*)d_in[5];   // [256][512]
    const float* eps_h = (const float*)d_in[6];   // [512][64][256]
    const float* eps_c = (const float*)d_in[7];   // [512][64][256]
    float* out = (float*)d_out;

    // Workspace layout
    float*  xW    = (float*)d_ws;                         // 32768*1024 fp32 = 128 MB
    __bf16* WihT  = (__bf16*)(xW + (size_t)MROWS * NGATE);
    __bf16* WhhT  = WihT + (size_t)NGATE * HSZ;
    __bf16* HrepT = WhhT + (size_t)NGATE * HSZ;
    __bf16* CrepT = HrepT + (size_t)NREP * HSZ;

    // Phase 1: transpose + convert weights to bf16 [N][K]
    {
        int n1 = HSZ * NGATE;
        transpose_cvt_kernel<<<(n1 + 255) / 256, 256, 0, stream>>>(w_ih, WihT, HSZ, NGATE);
        transpose_cvt_kernel<<<(n1 + 255) / 256, 256, 0, stream>>>(w_hh, WhhT, HSZ, NGATE);
        int n2 = HSZ * NREP;
        transpose_cvt_kernel<<<(n2 + 255) / 256, 256, 0, stream>>>(h_rep, HrepT, HSZ, NREP);
        transpose_cvt_kernel<<<(n2 + 255) / 256, 256, 0, stream>>>(c_rep, CrepT, HSZ, NREP);
    }

    // Phase 2: xW = X @ W_ih   (32768 x 256 x 1024)
    {
        // 2048 row-blocks * 16 col-groups = 32768 wave tasks; 8 waves/block
        gemm_xw_kernel<<<4096, 256, 0, stream>>>(x, WihT, xW);
    }

    // Phase 3: persistent recurrence, single WGP, 224 KB dynamic LDS
    {
        const int smem_bytes = BSZ * HSZ * 2 * 3 + BSZ * HSZ * 4 * 2;  // 229376
        hipFuncSetAttribute((const void*)lstm_recurrent_kernel,
                            hipFuncAttributeMaxDynamicSharedMemorySize, smem_bytes);
        lstm_recurrent_kernel<<<1, 1024, smem_bytes, stream>>>(
            xW, bias, WhhT, HrepT, CrepT, eps_h, eps_c, out);
    }
}
